// TemporalMultiheadAttention_84378927497476
// MI455X (gfx1250) — matrix operand order repaired and checked by reading, not compiled, not run on hardware
//
#include <hip/hip_runtime.h>
#include <hip/hip_bf16.h>

#define L_DIM 1024
#define S_DIM 1024
#define N_DIM 8
#define E_DIM 1024
#define H_DIM 16
#define HD    64

typedef __attribute__((ext_vector_type(16))) __bf16 v16bf;
typedef __attribute__((ext_vector_type(8)))  float  v8f;

__device__ __forceinline__ __bf16 f2b(float f) { return (__bf16)f; }

__device__ __forceinline__ v8f wmma_bf16(v16bf a, v16bf b, v8f c) {
    // D = A(16x32) * B(32x16) + C, f32 accumulate
    return __builtin_amdgcn_wmma_f32_16x16x32_bf16(false, a, false, b, (short)0, c, false, false);
}

// ---------------------------------------------------------------------------
// Kernel 1: fused QKV projection, LDS-staged.
// Workgroup (256 thr, 8 waves) computes a 128x128 tile of C = X . W^T + bias.
// K=1024 staged in 32-wide bf16 chunks (16 KB LDS); each wave owns a 16-row
// strip and 8 WMMA accumulators.  blockIdx.z = 0/1/2 -> q/k/v.
// Outputs scattered to per-head layouts (bf16); q scaled by hd^-0.5.
// ---------------------------------------------------------------------------
__global__ __launch_bounds__(256) void qkv_proj_kernel(
    const float* __restrict__ qin, const float* __restrict__ kin,
    const float* __restrict__ vin, const float* __restrict__ w,
    const float* __restrict__ bias,
    __bf16* __restrict__ q_ws, __bf16* __restrict__ k_ws, __bf16* __restrict__ vT_ws)
{
    __shared__ __bf16 sA[128][32];   // 8 KB
    __shared__ __bf16 sB[128][32];   // 8 KB

    const int tid  = threadIdx.x;
    const int lane = tid & 31;
    const int wv   = tid >> 5;       // 0..7
    const int mode = blockIdx.z;
    const int m0   = blockIdx.x * 128;
    const int n0   = blockIdx.y * 128;
    const int hi   = lane >> 4;
    const int ln   = lane & 15;

    const float* A = (mode == 0) ? qin : (mode == 1) ? kin : vin;

    // staging assignment: 2 threads per row, 16 floats each
    const int sr = tid >> 1;
    const int sh = (tid & 1) * 16;
    const float* aload = A + (size_t)(m0 + sr) * E_DIM + sh;
    const float* bload = w + ((size_t)mode * E_DIM + n0 + sr) * E_DIM + sh;

    v8f acc[8];
    #pragma unroll
    for (int nt = 0; nt < 8; ++nt) acc[nt] = (v8f){};

    for (int k0 = 0; k0 < E_DIM; k0 += 32) {
        {   // stage chunk, fp32 -> bf16
            const float* ap = aload + k0;
            const float* bp = bload + k0;
            v16bf ta, tb;
            #pragma unroll
            for (int j = 0; j < 16; ++j) { ta[j] = f2b(ap[j]); tb[j] = f2b(bp[j]); }
            *(v16bf*)&sA[sr][sh] = ta;
            *(v16bf*)&sB[sr][sh] = tb;
            if (k0 + 32 < E_DIM) {
                __builtin_prefetch(ap + 32, 0, 0);   // global_prefetch_b8
                __builtin_prefetch(bp + 32, 0, 0);
            }
        }
        __syncthreads();

        v16bf a;
        const __bf16* arow = &sA[wv * 16 + ln][0];
        #pragma unroll
        for (int j = 0; j < 8; ++j) { a[j] = arow[hi * 8 + j]; a[j + 8] = arow[hi * 8 + 16 + j]; }
        #pragma unroll
        for (int nt = 0; nt < 8; ++nt) {
            v16bf b;
            const __bf16* brow = &sB[nt * 16 + ln][0];
            #pragma unroll
            for (int j = 0; j < 8; ++j) { b[j] = brow[hi * 8 + j]; b[j + 8] = brow[hi * 8 + 16 + j]; }
            acc[nt] = wmma_bf16(a, b, acc[nt]);
        }
        __syncthreads();
    }

    // epilogue: bias + scatter to per-head layouts
    #pragma unroll
    for (int nt = 0; nt < 8; ++nt) {
        const int eg   = n0 + nt * 16 + ln;
        const float bb = bias[mode * E_DIM + eg];
        const int hcol = eg >> 6;
        const int d    = eg & 63;
        #pragma unroll
        for (int i = 0; i < 8; ++i) {
            const int r  = m0 + wv * 16 + i + hi * 8;
            const int l  = r >> 3;     // sequence position (N=8)
            const int nb = r & 7;      // batch
            const float  val  = acc[nt][i] + bb;
            const size_t head = (size_t)(nb * H_DIM + hcol);
            if (mode == 0) {
                q_ws[(head * (size_t)L_DIM + l) * HD + d] = f2b(val * 0.125f);
            } else if (mode == 1) {
                k_ws[(head * (size_t)S_DIM + l) * HD + d] = f2b(val);
            } else {
                vT_ws[(head * (size_t)HD + d) * S_DIM + l] = f2b(val);   // transposed
            }
        }
    }
}

// ---------------------------------------------------------------------------
// Kernel 2: attention. One workgroup per (16 L-rows, batch nb); loop over heads.
// scores = (q.kT)*exp(t_l/TS)*exp(-t_s/TS) -> softmax -> PV, all via WMMA.
// attn_weights accumulated (sum_h attn / H) straight into owned d_out rows.
// ---------------------------------------------------------------------------
__global__ __launch_bounds__(512) void attn_kernel(
    const __bf16* __restrict__ q_ws, const __bf16* __restrict__ k_ws,
    const __bf16* __restrict__ vT_ws, const float* __restrict__ tvec,
    __bf16* __restrict__ ao_ws, float* __restrict__ aw_out)
{
    __shared__ float  sc[16][S_DIM];    // 64 KB score tile (f32)
    __shared__ __bf16 atn[16][S_DIM];   // 32 KB attn tile (bf16)
    __shared__ float  cf[S_DIM];        // 4 KB  col factors
    __shared__ float  rf[16];           // row factors

    const int lane = threadIdx.x & 31;
    const int wid  = threadIdx.x >> 5;  // 16 waves
    const int lt = blockIdx.x;          // 0..63
    const int nb = blockIdx.y;          // 0..7
    const int l0 = lt * 16;
    const int hi = lane >> 4;
    const int ln = lane & 15;

    for (int s = threadIdx.x; s < S_DIM; s += 512)
        cf[s] = __expf(-tvec[s * N_DIM + nb] * 1.0e-6f);
    if (threadIdx.x < 16)
        rf[threadIdx.x] = __expf(tvec[(l0 + threadIdx.x) * N_DIM + nb] * 1.0e-6f);
    __syncthreads();

    for (int h = 0; h < H_DIM; ++h) {
        const size_t head = (size_t)(nb * H_DIM + h);

        // ---- scores: 16 x 1024, K = 64 (two WMMA k-steps), 4 s-tiles per wave
        {
            const __bf16* qrow = q_ws + (head * L_DIM + l0 + ln) * HD;
            v16bf qa0, qa1;
            #pragma unroll
            for (int j = 0; j < 8; ++j) {
                qa0[j]     = qrow[hi * 8 + j];
                qa0[j + 8] = qrow[hi * 8 + 16 + j];
                qa1[j]     = qrow[32 + hi * 8 + j];
                qa1[j + 8] = qrow[32 + hi * 8 + 16 + j];
            }
            for (int st = wid; st < 64; st += 16) {
                const int s0 = st * 16;
                const __bf16* krow = k_ws + (head * S_DIM + s0 + ln) * HD;
                v16bf b0, b1;
                #pragma unroll
                for (int j = 0; j < 8; ++j) {
                    b0[j] = krow[hi * 8 + j]; b0[j + 8] = krow[hi * 8 + 16 + j];
                    b1[j] = krow[32 + hi * 8 + j]; b1[j + 8] = krow[32 + hi * 8 + 16 + j];
                }
                v8f acc = {};
                acc = wmma_bf16(qa0, b0, acc);
                acc = wmma_bf16(qa1, b1, acc);
                const int col = s0 + ln;
                const float cfv = cf[col];
                #pragma unroll
                for (int i = 0; i < 8; ++i) {
                    const int row = i + hi * 8;
                    sc[row][col] = acc[i] * rf[row] * cfv;
                }
            }
        }
        __syncthreads();

        // ---- softmax, one row per wave (wave32 shuffle reductions)
        {
            const int row = wid;
            float mx = -3.0e38f;
            for (int c = lane; c < S_DIM; c += 32) mx = fmaxf(mx, sc[row][c]);
            #pragma unroll
            for (int off = 16; off > 0; off >>= 1) mx = fmaxf(mx, __shfl_xor(mx, off, 32));
            float sum = 0.0f;
            for (int c = lane; c < S_DIM; c += 32) {
                const float p = __expf(sc[row][c] - mx);
                sc[row][c] = p;
                sum += p;
            }
            #pragma unroll
            for (int off = 16; off > 0; off >>= 1) sum += __shfl_xor(sum, off, 32);
            const float inv = 1.0f / sum;
            float* awrow = aw_out + ((size_t)nb * L_DIM + l0 + row) * S_DIM;
            for (int c = lane; c < S_DIM; c += 32) {
                const float p = sc[row][c] * inv;
                atn[row][c] = f2b(p);
                const float contrib = p * (1.0f / (float)H_DIM);
                if (h == 0) awrow[c] = contrib;     // first head overwrites poison
                else        awrow[c] += contrib;    // rows owned exclusively by this WG
            }
        }
        __syncthreads();

        // ---- PV: out16x64 = attn(16x1024) * v(1024x64); waves 0..3 take n-tiles
        if (wid < 4) {
            const int d0 = wid * 16;
            const __bf16* vrow = vT_ws + (head * HD + d0 + ln) * S_DIM;
            v8f acc = {};
            for (int kk = 0; kk < 32; ++kk) {
                const int k0 = kk * 32;
                v16bf a, b;
                const __bf16* ap = &atn[ln][k0 + hi * 8];
                #pragma unroll
                for (int j = 0; j < 8; ++j) { a[j] = ap[j]; a[j + 8] = ap[j + 16]; }
                const __bf16* bp = vrow + k0 + hi * 8;
                #pragma unroll
                for (int j = 0; j < 8; ++j) { b[j] = bp[j]; b[j + 8] = bp[j + 16]; }
                acc = wmma_bf16(a, b, acc);
            }
            #pragma unroll
            for (int i = 0; i < 8; ++i) {
                const int l = l0 + i + hi * 8;
                ao_ws[((size_t)l * N_DIM + nb) * E_DIM + h * HD + d0 + ln] = f2b(acc[i]);
            }
        }
        __syncthreads();
    }
}

// ---------------------------------------------------------------------------
// Kernel 3: output projection, LDS-staged (same tiling as kernel 1).
// out[r,e] = AO[r,:] . Wout[e,:] + bias, fp32 result.
// ---------------------------------------------------------------------------
__global__ __launch_bounds__(256) void out_proj_kernel(
    const __bf16* __restrict__ ao, const float* __restrict__ w,
    const float* __restrict__ bias, float* __restrict__ out)
{
    __shared__ __bf16 sA[128][32];
    __shared__ __bf16 sB[128][32];

    const int tid  = threadIdx.x;
    const int lane = tid & 31;
    const int wv   = tid >> 5;
    const int m0   = blockIdx.x * 128;
    const int n0   = blockIdx.y * 128;
    const int hi   = lane >> 4;
    const int ln   = lane & 15;

    const int sr = tid >> 1;
    const int sh = (tid & 1) * 16;
    const __bf16* aload = ao + (size_t)(m0 + sr) * E_DIM + sh;
    const float*  bload = w + (size_t)(n0 + sr) * E_DIM + sh;

    v8f acc[8];
    #pragma unroll
    for (int nt = 0; nt < 8; ++nt) acc[nt] = (v8f){};

    for (int k0 = 0; k0 < E_DIM; k0 += 32) {
        {
            const __bf16* ap = aload + k0;
            const float*  bp = bload + k0;
            v16bf tb;
            #pragma unroll
            for (int j = 0; j < 16; ++j) tb[j] = f2b(bp[j]);
            *(v16bf*)&sA[sr][sh] = *(const v16bf*)ap;   // bf16 passthrough
            *(v16bf*)&sB[sr][sh] = tb;
            if (k0 + 32 < E_DIM) {
                __builtin_prefetch(ap + 32, 0, 0);
                __builtin_prefetch(bp + 32, 0, 0);
            }
        }
        __syncthreads();

        v16bf a;
        const __bf16* arow = &sA[wv * 16 + ln][0];
        #pragma unroll
        for (int j = 0; j < 8; ++j) { a[j] = arow[hi * 8 + j]; a[j + 8] = arow[hi * 8 + 16 + j]; }
        #pragma unroll
        for (int nt = 0; nt < 8; ++nt) {
            v16bf b;
            const __bf16* brow = &sB[nt * 16 + ln][0];
            #pragma unroll
            for (int j = 0; j < 8; ++j) { b[j] = brow[hi * 8 + j]; b[j + 8] = brow[hi * 8 + 16 + j]; }
            acc[nt] = wmma_bf16(a, b, acc[nt]);
        }
        __syncthreads();
    }

    #pragma unroll
    for (int nt = 0; nt < 8; ++nt) {
        const int eg = n0 + nt * 16 + ln;
        const float bb = bias[eg];
        #pragma unroll
        for (int i = 0; i < 8; ++i)
            out[(size_t)(m0 + wv * 16 + i + hi * 8) * E_DIM + eg] = acc[nt][i] + bb;
    }
}

// ---------------------------------------------------------------------------
extern "C" void kernel_launch(void* const* d_in, const int* in_sizes, int n_in,
                              void* d_out, int out_size, void* d_ws, size_t ws_size,
                              hipStream_t stream)
{
    const float* query = (const float*)d_in[0];
    const float* key   = (const float*)d_in[1];
    const float* value = (const float*)d_in[2];
    const float* tvec  = (const float*)d_in[3];
    const float* in_w  = (const float*)d_in[4];
    const float* in_b  = (const float*)d_in[5];
    const float* out_w = (const float*)d_in[6];
    const float* out_b = (const float*)d_in[7];

    float* out = (float*)d_out;                               // (L,N,E)
    float* aw  = out + (size_t)L_DIM * N_DIM * E_DIM;         // (N,L,S)

    const size_t elems = (size_t)N_DIM * H_DIM * L_DIM * HD;  // 8,388,608
    __bf16* q_ws  = (__bf16*)d_ws;                            // 4 x 16 MiB bf16
    __bf16* k_ws  = q_ws  + elems;
    __bf16* vT_ws = k_ws  + elems;
    __bf16* ao_ws = vT_ws + elems;

    qkv_proj_kernel<<<dim3(64, 8, 3), 256, 0, stream>>>(query, key, value, in_w, in_b,
                                                        q_ws, k_ws, vT_ws);
    attn_kernel<<<dim3(64, 8), 512, 0, stream>>>(q_ws, k_ws, vT_ws, tvec, ao_ws, aw);
    out_proj_kernel<<<dim3(64, 8), 256, 0, stream>>>(ao_ws, out_w, out_b, out);
}
